// Atoms_49598282334696
// MI455X (gfx1250) — matrix operand order repaired and checked by reading, not compiled, not run on hardware
//
#include <hip/hip_runtime.h>
#include <hip/hip_bf16.h>
#include <cstddef>

// ---------------------------------------------------------------------------
// Pipeline (B=32, E=16, SIG=512 signals of 32768 samples):
//  k_params   : clip x, build per-signal scalars + res_mags + interpolated shape
//  k_tables   : f16 DFT tables in WMMA-fragment-major layout
//  k_fft_band : per-signal in-LDS 16384-pt complex FFT  -> rfft bins -> *shape
//               -> inverse pack -> in-LDS inverse FFT -> Gaussian envelope * amp
//  k_shift    : time-domain fractional circular delay (replaces 2x 65536-pt FFT)
//  k_fwd_dft  : WMMA GEMM  frames(128x512) x DFT(512x272) -> spec (f32 acc)
//  k_iir      : sequential resonance scan over 128 frames (coalesced layout)
//  k_inv_dft  : WMMA GEMM  spec(128x544) x IDFT(544x512) -> out_frames
//  k_final    : overlap-add + sum over events + per-batch max-norm
//
// spec layout is [frame][signal][bin] so the IIR scan is fully coalesced.
// DFT tables are stored fragment-major: for K-block kb and column n, the 16
// K-values lane (h,c) needs are contiguous -> single v16h (2x b128) load.
// Workspace budget ~305 MB (out_frames overlays the dead shape/atoms regions).
// ---------------------------------------------------------------------------

typedef _Float16 v16h __attribute__((ext_vector_type(16)));
typedef _Float16 v8h  __attribute__((ext_vector_type(8)));
typedef float    v8f  __attribute__((ext_vector_type(8)));

#define SIG        512          // B*E
#define NS         32768
#define NC_BIG     16385
#define NFRM       128
#define WIN        512
#define NBIN       257
#define NPAD       272          // 257 padded to 17 tiles of 16
#define KINV       544          // 2*257=514 padded to 17 k-steps of 32
#define SPSTRIDE   (SIG * NBIN) // per-frame stride in spec arrays

// ---------------------------------------------------------------- params ----
__global__ __launch_bounds__(256)
void k_params(const float* __restrict__ x, float* __restrict__ shape,
              float* __restrict__ resM, float4* __restrict__ prm)
{
    const int s = blockIdx.x, tid = threadIdx.x;
    const float* row = x + (size_t)s * 388;
    if (tid == 0) {
        float means = __saturatef(row[0]);
        float stds  = __saturatef(row[1]);
        float amp   = __saturatef(row[387]);
        float sigma = fminf((1e-8f + stds) * 32768.f, 32767.f);
        float mm    = 1.f / (sigma * 2.50662827f);          // max of gaussian pdf
        float4 P;
        P.x = mm / (mm + 1e-8f);                            // envelope scale
        P.y = 0.5f / (sigma * sigma);                       // 1/(2 sigma^2)
        P.z = amp;
        P.w = means * 32768.f * (65536.f / 32769.f);        // effective delay
        prm[s] = P;
    }
    for (int i = tid; i < NBIN; i += 256)
        resM[s * NBIN + i] = 0.01f + 0.99f * __saturatef(row[2 + i]);
    float* sh = shape + (size_t)s * NC_BIG;
    for (int i = tid; i < NC_BIG; i += 256) {
        float pos = ((float)i + 0.5f) * (128.f / 16385.f) - 0.5f;
        pos = fminf(fmaxf(pos, 0.f), 127.f);
        int   i0 = (int)floorf(pos);
        int   i1 = min(i0 + 1, 127);
        float w  = pos - (float)i0;
        float s0 = __saturatef(row[259 + i0]);
        float s1 = __saturatef(row[259 + i1]);
        sh[i] = s0 * (1.f - w) + s1 * w;
    }
}

// ---------------------------------------------------------------- tables ----
// Fragment-major layout:
//   fwd :  tab[((kb*NPAD + n)*32) + h*16 + i]  holds  W[k = kb*32 + i + 16h][n]
//   inv : itab[((kb*WIN  + j)*32) + h*16 + i]  holds IW[kk= kb*32 + i + 16h][j]
// so lane (h, c=n%16) of a wave loads its whole 16-half B fragment contiguously.
__global__ __launch_bounds__(256)
void k_tables(_Float16* __restrict__ Ctab, _Float16* __restrict__ Stab,
              _Float16* __restrict__ ITab)
{
    const int idx = blockIdx.x * blockDim.x + threadIdx.x;
    const float inv = 0.04419417382f;                       // 1/sqrt(512)
    if (idx < 16 * NPAD * 32) {                             // forward rDFT (K=512)
        int kb = idx / (NPAD * 32);
        int rem = idx % (NPAD * 32);
        int n = rem >> 5, t = rem & 31;
        int h = t >> 4, i = t & 15;
        int k = kb * 32 + i + 16 * h;
        float ang = 6.283185307f * (float)((k * n) % 512) / 512.f;
        float sn, cs; __sincosf(ang, &sn, &cs);
        Ctab[idx] = (_Float16)(cs * inv);
        Stab[idx] = (_Float16)(-sn * inv);
    }
    if (idx < 17 * WIN * 32) {                              // inverse rDFT (K=544)
        int kb = idx / (WIN * 32);
        int rem = idx % (WIN * 32);
        int j = rem >> 5, t = rem & 31;
        int h = t >> 4, i = t & 15;
        int kk = kb * 32 + i + 16 * h;                      // 2k = Re row, 2k+1 = Im row
        float v = 0.f;
        if (kk < 514) {
            int   k  = kk >> 1;
            float wk = (k == 0 || k == 256) ? 1.f : 2.f;
            float ang = 6.283185307f * (float)((k * j) % 512) / 512.f;
            float sn, cs; __sincosf(ang, &sn, &cs);
            v = (kk & 1) ? (-wk * sn * inv) : (wk * cs * inv);
        }
        ITab[idx] = (_Float16)v;
    }
}

// ------------------------------------------------- in-LDS band-pass FFT ----
// 16384-pt complex FFT of packed real signal, fully resident in 128 KB LDS
// (CDNA5 WGP has 320 KB). Forward -> rfft split -> *shape -> inverse pack ->
// inverse FFT -> gaussian envelope * amp.
__global__ __launch_bounds__(1024)
void k_fft_band(const float* __restrict__ noise, const float* __restrict__ shape,
                const float4* __restrict__ prm, float* __restrict__ atoms)
{
    extern __shared__ float lds[];
    float* RE = lds;
    float* IM = lds + 16384;
    const int s = blockIdx.x, tid = threadIdx.x;
    const float* np = noise + (size_t)s * NS;

    // load packed, bit-reversed (14-bit)
    for (int i = tid; i < 16384; i += 1024) {
        __builtin_prefetch(np + 2 * i + 4096, 0, 1);        // global_prefetch_b8
        int r = __brev((unsigned)i) >> 18;
        RE[r] = np[2 * i];
        IM[r] = np[2 * i + 1];
    }
    __syncthreads();

    // forward radix-2 DIT stages
    for (int st = 1; st <= 14; ++st) {
        const int m = 1 << st, h = m >> 1;
        for (int j = tid; j < 8192; j += 1024) {
            int grp = j >> (st - 1);
            int pos = j & (h - 1);
            int i0  = (grp << st) + pos, i1 = i0 + h;
            float ang = -6.283185307f * (float)pos / (float)m;
            float sn, cs; __sincosf(ang, &sn, &cs);
            float r1 = RE[i1], q1 = IM[i1];
            float tr = cs * r1 - sn * q1;
            float ti = cs * q1 + sn * r1;
            float r0 = RE[i0], q0 = IM[i0];
            RE[i1] = r0 - tr; IM[i1] = q0 - ti;
            RE[i0] = r0 + tr; IM[i0] = q0 + ti;
        }
        __syncthreads();
    }

    // rfft split  ->  * shape  ->  inverse real pack (all in place, per pair)
    const float sc = 0.0055242717f;                         // 1/sqrt(32768)
    const float* sh = shape + (size_t)s * NC_BIG;
    for (int k = tid; k <= 8192; k += 1024) {
        if (k == 0) {
            float R = RE[0], I = IM[0];
            float Y0 = (R + I) * sc * sh[0];
            float YN = (R - I) * sc * sh[16384];
            RE[0] = 0.5f * (Y0 + YN);
            IM[0] = 0.5f * (Y0 - YN);
        } else {
            int nk = 16384 - k;                             // k==8192 -> self pair
            float akr = RE[k],  aki = IM[k];
            float bkr = RE[nk], bki = IM[nk];
            float ZeR = 0.5f * (akr + bkr), ZeI = 0.5f * (aki - bki);
            float ZoR = 0.5f * (aki + bki), ZoI = -0.5f * (akr - bkr);
            float cp, sp; __sincosf(3.14159265f * (float)k / 16384.f, &sp, &cp);
            float XkR = ZeR + cp * ZoR + sp * ZoI;
            float XkI = ZeI + cp * ZoI - sp * ZoR;
            float XnR = ZeR - cp * ZoR - sp * ZoI;
            float XnI = -ZeI + cp * ZoI - sp * ZoR;
            float g0 = sc * sh[k], g1 = sc * sh[nk];
            float YkR = XkR * g0, YkI = XkI * g0;
            float YnR = XnR * g1, YnI = XnI * g1;
            float YeR = 0.5f * (YkR + YnR), YeI = 0.5f * (YkI - YnI);
            float DR  = YkR - YnR,          DI  = YkI + YnI;
            float YoR = 0.5f * (cp * DR - sp * DI);
            float YoI = 0.5f * (cp * DI + sp * DR);
            RE[k] = YeR - YoI; IM[k] = YeI + YoR;
            if (k < 8192) { RE[nk] = YeR + YoI; IM[nk] = -YeI + YoR; }
        }
    }
    __syncthreads();

    // bit-reverse permute back, then inverse stages (+ twiddle sign)
    for (int i = tid; i < 16384; i += 1024) {
        int r = __brev((unsigned)i) >> 18;
        if (r > i) {
            float tr = RE[i], ti = IM[i];
            RE[i] = RE[r]; IM[i] = IM[r];
            RE[r] = tr;    IM[r] = ti;
        }
    }
    __syncthreads();
    for (int st = 1; st <= 14; ++st) {
        const int m = 1 << st, h = m >> 1;
        for (int j = tid; j < 8192; j += 1024) {
            int grp = j >> (st - 1);
            int pos = j & (h - 1);
            int i0  = (grp << st) + pos, i1 = i0 + h;
            float ang = 6.283185307f * (float)pos / (float)m;
            float sn, cs; __sincosf(ang, &sn, &cs);
            float r1 = RE[i1], q1 = IM[i1];
            float tr = cs * r1 - sn * q1;
            float ti = cs * q1 + sn * r1;
            float r0 = RE[i0], q0 = IM[i0];
            RE[i1] = r0 - tr; IM[i1] = q0 - ti;
            RE[i0] = r0 + tr; IM[i0] = q0 + ti;
        }
        __syncthreads();
    }

    // unpack, scale, gaussian envelope, amp
    const float4 P  = prm[s];
    const float  so = 0.011048543f;                         // sqrt(32768)/16384
    float* ap = atoms + (size_t)s * NS;
    for (int i = tid; i < 16384; i += 1024) {
        float t0 = (float)(2 * i), t1 = (float)(2 * i + 1);
        float e0 = P.x * __expf(-P.y * t0 * t0);
        float e1 = P.x * __expf(-P.y * t1 * t1);
        ap[2 * i]     = RE[i] * so * e0 * P.z;
        ap[2 * i + 1] = IM[i] * so * e1 * P.z;
    }
}

// ------------------------------------------------ fractional delay shift ----
__global__ __launch_bounds__(256)
void k_shift(const float* __restrict__ atoms, const float4* __restrict__ prm,
             float* __restrict__ atoms2)
{
    long long g = (long long)blockIdx.x * blockDim.x + threadIdx.x;
    if (g >= (long long)SIG * NS) return;
    int s = (int)(g >> 15), n = (int)(g & (NS - 1));
    float d   = prm[s].w;
    float src = (float)n - d;
    src = src - 65536.f * floorf(src * (1.f / 65536.f));
    int   i0 = (int)floorf(src);
    float w  = src - (float)i0;
    const float* ap = atoms + (size_t)s * NS;
    int a0 = i0 & 65535, a1 = (i0 + 1) & 65535;
    float v0 = (a0 < NS) ? ap[a0] : 0.f;                    // zero-padded half
    float v1 = (a1 < NS) ? ap[a1] : 0.f;
    atoms2[(size_t)s * NS + n] = v0 * (1.f - w) + v1 * w;
}

// ------------------------------------------- forward frame DFT via WMMA ----
// block = 128 threads (4 waves); one (signal, 16-frame M-tile) per block.
// A fragment = 2x ds_load_b128 ; B fragment = contiguous v16h from global.
__global__ __launch_bounds__(128)
void k_fwd_dft(const float* __restrict__ atoms2, const _Float16* __restrict__ Ctab,
               const _Float16* __restrict__ Stab, float* __restrict__ specRe,
               float* __restrict__ specIm)
{
    __shared__ _Float16 Afrm[16 * WIN];                     // windowed frames, f16
    const int s = blockIdx.x >> 3, mtile = blockIdx.x & 7, tid = threadIdx.x;
    const float* ap = atoms2 + (size_t)s * NS;
    for (int e = tid; e < 16 * WIN; e += 128) {
        int r = e >> 9, j = e & 511;
        int idx = (mtile * 16 + r) * 256 + j;
        float v = (idx < NS) ? ap[idx] : 0.f;               // STEP zero-pad tail
        float w = 0.54f - 0.46f * __cosf(6.283185307f * (float)j / 512.f);
        Afrm[e] = (_Float16)(v * w);
    }
    __syncthreads();

    const int wave = tid >> 5, lane = tid & 31;
    const int hh = lane >> 4, cc = lane & 15;
    const v8h* arow = (const v8h*)(Afrm + cc * WIN);        // 8-half chunks
    for (int nt = wave; nt < 17; nt += 4) {
        const int n0 = nt * 16;
        v8f accR = {}; v8f accI = {};
        for (int k0 = 0; k0 < WIN; k0 += 32) {
            // A: halves [k0+16h .. +15] of row cc
            v8h lo = arow[(k0 >> 3) + 2 * hh];
            v8h hi = arow[(k0 >> 3) + 2 * hh + 1];
            v16h a = __builtin_shufflevector(lo, hi, 0, 1, 2, 3, 4, 5, 6, 7,
                                             8, 9, 10, 11, 12, 13, 14, 15);
            // B: fragment-major tables, one v16h load each
            size_t bo = ((size_t)(k0 >> 5) * NPAD + (n0 + cc)) * 32 + hh * 16;
            v16h bR = *(const v16h*)(Ctab + bo);
            v16h bI = *(const v16h*)(Stab + bo);
            accR = __builtin_amdgcn_wmma_f32_16x16x32_f16(false, a, false, bR,
                                                          (short)0, accR, false, false);
            accI = __builtin_amdgcn_wmma_f32_16x16x32_f16(false, a, false, bI,
                                                          (short)0, accI, false, false);
        }
        const int bin = n0 + cc;
#pragma unroll
        for (int r = 0; r < 8; ++r) {
            int f = mtile * 16 + r + 8 * hh;
            if (bin < NBIN) {                               // spec[frame][signal][bin]
                size_t o = (size_t)f * SPSTRIDE + s * NBIN + bin;
                specRe[o] = accR[r];
                specIm[o] = accI[r];
            }
        }
    }
}

// ----------------------------------------------------- resonance IIR scan ----
// spec is [frame][signal*bin] -> each step is a fully coalesced 512KB slice.
__global__ __launch_bounds__(256)
void k_iir(float* __restrict__ specRe, float* __restrict__ specIm,
           const float* __restrict__ resM)
{
    int t = blockIdx.x * blockDim.x + threadIdx.x;
    if (t >= SIG * NBIN) return;
    float rm = resM[t];
    float pr = specRe[t], pi = specIm[t];                   // scan init = frame 0
    for (int f = 0; f < NFRM; ++f) {
        size_t o = (size_t)f * SPSTRIDE + t;
        float cr = specRe[o] + pr * rm;
        float ci = specIm[o] + pi * rm;
        specRe[o] = cr; specIm[o] = ci;
        pr = cr; pi = ci;
    }
}

// ------------------------------------------ inverse frame DFT via WMMA ----
__global__ __launch_bounds__(128)
void k_inv_dft(const float* __restrict__ specRe, const float* __restrict__ specIm,
               const _Float16* __restrict__ ITab, float* __restrict__ outF)
{
    __shared__ _Float16 Acat[16 * KINV];                    // [frame][2k|2k+1]
    const int s = blockIdx.x >> 3, mtile = blockIdx.x & 7, tid = threadIdx.x;
    for (int e = tid; e < 16 * KINV; e += 128) {
        int r = e / KINV, kk = e % KINV;
        float v = 0.f;
        if (kk < 514) {
            int    k = kk >> 1;
            size_t o = (size_t)(mtile * 16 + r) * SPSTRIDE + s * NBIN + k;
            v = (kk & 1) ? specIm[o] : specRe[o];
        }
        Acat[e] = (_Float16)v;
    }
    __syncthreads();

    const int wave = tid >> 5, lane = tid & 31;
    const int hh = lane >> 4, cc = lane & 15;
    const v8h* arow = (const v8h*)(Acat + cc * KINV);
    for (int nt = wave; nt < 32; nt += 4) {
        const int n0 = nt * 16;
        v8f acc = {};
        for (int k0 = 0; k0 < KINV; k0 += 32) {
            v8h lo = arow[(k0 >> 3) + 2 * hh];
            v8h hi = arow[(k0 >> 3) + 2 * hh + 1];
            v16h a = __builtin_shufflevector(lo, hi, 0, 1, 2, 3, 4, 5, 6, 7,
                                             8, 9, 10, 11, 12, 13, 14, 15);
            size_t bo = ((size_t)(k0 >> 5) * WIN + (n0 + cc)) * 32 + hh * 16;
            v16h b = *(const v16h*)(ITab + bo);
            acc = __builtin_amdgcn_wmma_f32_16x16x32_f16(false, a, false, b,
                                                         (short)0, acc, false, false);
        }
#pragma unroll
        for (int r = 0; r < 8; ++r) {
            int f = mtile * 16 + r + 8 * hh;
            outF[((size_t)s * NFRM + f) * WIN + n0 + cc] = acc[r];
        }
    }
}

// -------------------------------- overlap-add + event sum + max-norm ----
__global__ __launch_bounds__(1024)
void k_final(const float* __restrict__ outF, float* __restrict__ out)
{
    __shared__ float red[1024];
    const int b = blockIdx.x, tid = threadIdx.x;
    float vals[32];
    float vmax = 0.f;
#pragma unroll
    for (int it = 0; it < 32; ++it) {
        int n = tid + it * 1024;
        int q = n >> 8, j = n & 255;
        float v = 0.f;
        for (int e = 0; e < 16; ++e) {
            const float* OF = outF + ((size_t)(b * 16 + e) * NFRM) * WIN;
            v += OF[q * WIN + j];                           // first halves
            if (q > 0) v += OF[(q - 1) * WIN + 256 + j];    // second halves, +STEP
        }
        vals[it] = v;
        vmax = fmaxf(vmax, fabsf(v));
    }
    red[tid] = vmax;
    __syncthreads();
    for (int st = 512; st > 0; st >>= 1) {
        if (tid < st) red[tid] = fmaxf(red[tid], red[tid + st]);
        __syncthreads();
    }
    const float inv = 1.f / (red[0] + 1e-8f);
#pragma unroll
    for (int it = 0; it < 32; ++it)
        out[(size_t)b * NS + tid + it * 1024] = vals[it] * inv;
}

// --------------------------------------------------------------- launch ----
extern "C" void kernel_launch(void* const* d_in, const int* in_sizes, int n_in,
                              void* d_out, int out_size, void* d_ws, size_t ws_size,
                              hipStream_t stream)
{
    (void)in_sizes; (void)n_in; (void)out_size; (void)ws_size;
    const float* x     = (const float*)d_in[0];
    const float* noise = (const float*)d_in[1];
    float*       out   = (float*)d_out;

    char* w = (char*)d_ws;
    const size_t SH_B = (size_t)SIG * NC_BIG * 4;           //  33,556,480
    const size_t AT_B = (size_t)SIG * NS * 4;               //  67,108,864
    const size_t SP_B = (size_t)SIG * NBIN * NFRM * 4;      //  67,371,008
    float* shape  = (float*)(w);
    float* atoms  = (float*)(w + SH_B);
    float* atoms2 = (float*)(w + SH_B + AT_B);
    float* outF   = (float*)(w);                            // overlays dead regions
    size_t off    = SH_B + 2 * AT_B;                        // 167,774,208 (>= outF's 134 MB)
    float* specRe = (float*)(w + off); off += SP_B;
    float* specIm = (float*)(w + off); off += SP_B;
    _Float16* Ctab = (_Float16*)(w + off); off += (size_t)16 * NPAD * 32 * 2;
    _Float16* Stab = (_Float16*)(w + off); off += (size_t)16 * NPAD * 32 * 2;
    _Float16* ITab = (_Float16*)(w + off); off += (size_t)17 * WIN * 32 * 2;
    float*  resM   = (float*)(w + off);    off += (size_t)SIG * NBIN * 4;
    float4* prm    = (float4*)(w + off);   off += (size_t)SIG * 16;

    // allow 128 KB dynamic LDS for the resident FFT (320 KB/WGP on CDNA5)
    (void)hipFuncSetAttribute(reinterpret_cast<const void*>(k_fft_band),
                              hipFuncAttributeMaxDynamicSharedMemorySize, 131072);

    k_params <<<SIG, 256, 0, stream>>>(x, shape, resM, prm);
    k_tables <<<(17 * WIN * 32 + 255) / 256, 256, 0, stream>>>(Ctab, Stab, ITab);
    k_fft_band<<<SIG, 1024, 131072, stream>>>(noise, shape, prm, atoms);
    k_shift  <<<(SIG * NS) / 256, 256, 0, stream>>>(atoms, prm, atoms2);
    k_fwd_dft<<<SIG * 8, 128, 0, stream>>>(atoms2, Ctab, Stab, specRe, specIm);
    k_iir    <<<(SIG * NBIN + 255) / 256, 256, 0, stream>>>(specRe, specIm, resM);
    k_inv_dft<<<SIG * 8, 128, 0, stream>>>(specRe, specIm, ITab, outF);
    k_final  <<<32, 1024, 0, stream>>>(outF, out);
}
